// Model_9835475108474
// MI455X (gfx1250) — matrix operand order repaired and checked by reading, not compiled, not run on hardware
//
#include <hip/hip_runtime.h>
#include <hip/hip_bf16.h>
#include <math.h>

#if __has_include(<hip/amd_detail/amd_gfx1250_TDM.h>)
#define TDM_6ARG 1
#endif

#define NG     4096            // B*SEQ graphs
#define AN     22              // nodes per graph
#define FINF   4
#define HID    128
#define NHEAD  4
#define SEQL   128
#define NBATCH 32
#define DFFD   256
#define GHD    (AN*HID)        // 2816
#define NROW   (NG*AN)         // 90112
#define NPROJ  (NHEAD*HID)     // 512
#define NOUT   1408            // PRED*COUT
#define EPSF   1e-5f

#define KCHUNK 256             // K elements staged per TDM chunk
#define NBLK   64              // N columns per workgroup (B panel width)
#define MBLK   128             // M rows per workgroup (8 waves x 16)

typedef __attribute__((ext_vector_type(16))) _Float16 v16h;
typedef __attribute__((ext_vector_type(8)))  _Float16 v8h;
typedef __attribute__((ext_vector_type(8)))  float    v8f;
typedef unsigned int u32x4 __attribute__((ext_vector_type(4)));
typedef int          i32x4 __attribute__((ext_vector_type(4)));
typedef int          i32x8 __attribute__((ext_vector_type(8)));

__device__ __forceinline__ float sigmoidf_(float x){ return 1.f/(1.f+__expf(-x)); }

// ---- Tensor Data Mover: 2D panel load (NBLK rows x `chunk` f16) from BT (row stride K) ----
__device__ __forceinline__ void tdm_load_b_panel(const _Float16* gsrc, unsigned lds_addr,
                                                 int chunk, int K){
  unsigned long long ga = (unsigned long long)(size_t)gsrc;
  // D# group0: count=1 | lds_addr | global_addr[56:0] | type=2
  u32x4 g0;
  g0[0] = 1u;
  g0[1] = lds_addr;
  g0[2] = (unsigned)(ga & 0xffffffffu);
  g0[3] = (unsigned)((ga >> 32) & 0x1ffffffu) | (2u << 30);
  // D# group1: wg_mask=0, data_size=1(2B), tensor_dim0=K, tensor_dim1=NBLK,
  //            tile_dim0=chunk, tile_dim1=NBLK, tensor_dim0_stride=K
  unsigned td0 = (unsigned)K, td1 = (unsigned)NBLK;
  unsigned long long st0 = (unsigned long long)(unsigned)K;
  i32x8 g1;
  g1[0] = (int)(1u << 16);                                        // data_size
  g1[1] = (int)((td0 & 0xffffu) << 16);                           // tensor_dim0 lo
  g1[2] = (int)((td0 >> 16) | ((td1 & 0xffffu) << 16));           // tensor_dim0 hi | tensor_dim1 lo
  g1[3] = (int)((td1 >> 16) | (((unsigned)chunk & 0xffffu) << 16));// tensor_dim1 hi | tile_dim0
  g1[4] = (int)(unsigned)NBLK;                                    // tile_dim1 (tile_dim2=0)
  g1[5] = (int)(unsigned)(st0 & 0xffffffffu);                     // stride0 [31:0]
  g1[6] = (int)(unsigned)((st0 >> 32) & 0xffffu);                 // stride0 [47:32]
  g1[7] = 0;
  i32x4 z4 = {0,0,0,0};
#if defined(TDM_6ARG)
  i32x8 z8 = {0,0,0,0,0,0,0,0};
  __builtin_amdgcn_tensor_load_to_lds(g0, g1, z4, z4, z8, 0);
#else
  __builtin_amdgcn_tensor_load_to_lds(g0, g1, z4, z4, 0);
#endif
}

// ---------------- weight conversion ----------------
__global__ void k_cvt_f16(const float* __restrict__ src, _Float16* __restrict__ dst, int n){
  int i = blockIdx.x*blockDim.x + threadIdx.x;
  if (i < n) dst[i] = (_Float16)src[i];
}
// src (R,C) f32 row-major -> dst (C,R) f16 row-major
__global__ void k_transpose_f16(const float* __restrict__ src, _Float16* __restrict__ dst, int R, int C){
  int i = blockIdx.x*blockDim.x + threadIdx.x;
  if (i < R*C){ int r = i / C, c = i % C; dst[c*R + r] = (_Float16)src[i]; }
}

// ---------------- adjacency mask (one wave32 per graph) ----------------
__global__ void k_mask(const float* __restrict__ xe, unsigned char* __restrict__ mask){
  __shared__ float sx[AN], sy[AN];
  __shared__ float sadj[AN*AN];
  __shared__ float spart[32];
  int g = blockIdx.x, t = threadIdx.x;
  if (t < AN){
    const float* xp = xe + (size_t)g*(AN*FINF) + t*FINF;
    sx[t] = xp[0]; sy[t] = xp[1];
  }
  __syncthreads();
  float lsum = 0.f;
  for (int p = t; p < AN*AN; p += 32){
    int s = p / AN, u = p % AN;
    float dx = sx[s]-sx[u], dy = sy[s]-sy[u];
    float d  = sqrtf(fmaxf(dx*dx + dy*dy, 0.f));
    sadj[p] = d; lsum += d;
  }
  spart[t] = lsum; __syncthreads();
  if (t == 0){ float tot=0.f; for (int i=0;i<32;i++) tot += spart[i]; spart[0] = tot/(float)(AN*AN); }
  __syncthreads();
  float sigma = spart[0];
  float inv = 1.f/(2.f*sigma*sigma);
  __syncthreads();
  for (int p = t; p < AN*AN; p += 32){ float d = sadj[p]; sadj[p] = __expf(-d*d*inv); }
  __syncthreads();
  if (t < AN){                     // k-th (k=8) order statistic per row
    float rv[AN];
    for (int j=0;j<AN;j++) rv[j] = sadj[t*AN+j];
    float thr = 0.f;
    for (int it=0; it<8; ++it){
      int bi=0; float bv=-1e30f;
      for (int j=0;j<AN;j++) if (rv[j] > bv){ bv = rv[j]; bi = j; }
      thr = bv; rv[bi] = -1e30f;
    }
    unsigned char* mrow = mask + (size_t)g*(AN*AN) + (size_t)t*AN;
    for (int j=0;j<AN;j++) mrow[j] = (sadj[t*AN+j] >= thr) ? 1 : 0;
  }
}

// ---------------- GAT layer-0 projections (K=4, plain VALU) ----------------
__global__ void k_gat0_proj(const float* __restrict__ xe,
                            const float* __restrict__ Wl, const float* __restrict__ bl,
                            const float* __restrict__ Wr, const float* __restrict__ br,
                            float* __restrict__ XL, float* __restrict__ XR){
  size_t i = (size_t)blockIdx.x*blockDim.x + threadIdx.x;
  if (i >= (size_t)NROW*NPROJ) return;
  int    c   = (int)(i % NPROJ);
  size_t row = i / NPROJ;
  int    a   = (int)(row % AN);
  size_t g   = row / AN;
  const float* xp = xe + g*(AN*FINF) + (size_t)a*FINF;
  float accl = bl[c], accr = br[c];
  #pragma unroll
  for (int f=0; f<FINF; ++f){ float xv = xp[f]; accl += xv*Wl[f*NPROJ+c]; accr += xv*Wr[f*NPROJ+c]; }
  XL[i] = accl; XR[i] = accr;
}

// ------------- WMMA GEMM: C(M,N)=A(M,K)*B + bias, BT=(N,K) f16 -------------
// Workgroup: 128x64 tile of C; 8 waves, each a 16x64 strip (A frag reused x4).
// B panel (64 x KCHUNK) staged in LDS by the Tensor Data Mover, double-buffered
// on TENSORcnt; WMMA B-fragments come from LDS.
__global__ void k_wmma_gemm(const _Float16* __restrict__ A, const _Float16* __restrict__ BT,
                            const float* __restrict__ bias, float* __restrict__ C,
                            int M, int N, int K){
  __shared__ __align__(16) _Float16 sB[2][NBLK*KCHUNK];   // 2 x 32 KB
  int lane = threadIdx.x & 31;
  int wave = threadIdx.x >> 5;
  int n0 = blockIdx.x * NBLK;
  int m0 = blockIdx.y * MBLK + wave*16;
  bool active = (m0 < M);
  int rowA = active ? (m0 + (lane & 15)) : 0;
  int ka = (lane >> 4) * 8;      // A frag (16b A 16x32 ISA layout)
  int kb = (lane >> 4) * 8;      // B frag: col per lane, same K order as A
  const _Float16* pa = A + (size_t)rowA*K + ka;
  v8f acc[4] = {};
  int nchunks = (K + KCHUNK - 1)/KCHUNK;
  if (wave == 0){
    int c0 = (K < KCHUNK) ? K : KCHUNK;
    tdm_load_b_panel(BT + (size_t)n0*K, (unsigned)(size_t)&sB[0][0], c0, K);
  }
  for (int c = 0; c < nchunks; ++c){
    int k0 = c*KCHUNK;
    int chunk = (K - k0 < KCHUNK) ? (K - k0) : KCHUNK;
    if (wave == 0){
      if (c + 1 < nchunks){
        int k1 = k0 + KCHUNK;
        int chunk1 = (K - k1 < KCHUNK) ? (K - k1) : KCHUNK;
        tdm_load_b_panel(BT + (size_t)n0*K + k1, (unsigned)(size_t)&sB[(c+1)&1][0], chunk1, K);
        __builtin_amdgcn_s_wait_tensorcnt(1);   // chunk c complete (in-order)
      } else {
        __builtin_amdgcn_s_wait_tensorcnt(0);
      }
    }
    __syncthreads();
    if (active){
      const _Float16* sb = &sB[c&1][0];
      for (int kk = 0; kk < chunk; kk += 32){
        __builtin_prefetch(pa + k0 + kk + 128, 0, 0);     // global_prefetch_b8
        v8h alo = *(const v8h*)(pa + k0 + kk);
        v8h ahi = *(const v8h*)(pa + k0 + kk + 16);
        v16h af;
        #pragma unroll
        for (int i=0;i<8;i++){ af[i]=alo[i]; af[i+8]=ahi[i]; }
        #pragma unroll
        for (int nt=0; nt<4; ++nt){
          const _Float16* pb = sb + (size_t)(nt*16 + (lane & 15))*chunk + kk + kb;
          v8h blo = *(const v8h*)(pb);
          v8h bhi = *(const v8h*)(pb + 16);
          v16h bf;
          #pragma unroll
          for (int i=0;i<8;i++){ bf[i]=blo[i]; bf[i+8]=bhi[i]; }
          acc[nt] = __builtin_amdgcn_wmma_f32_16x16x32_f16(false, af, false, bf,
                                                           (short)0, acc[nt], false, false);
        }
      }
    }
    __syncthreads();
  }
  if (active){
    int rbase = m0 + ((lane>>4)<<3);
    #pragma unroll
    for (int nt=0; nt<4; ++nt){
      int co = n0 + nt*16 + (lane & 15);
      float bv = bias ? bias[co] : 0.f;
      #pragma unroll
      for (int r=0;r<8;r++) C[(size_t)(rbase+r)*N + co] = acc[nt][r] + bv;
    }
  }
}

// ---------------- GATv2 attention + aggregation + LN + relu (+residual), one block/graph ----------------
__global__ void k_gat_attn(const float* __restrict__ XL, const float* __restrict__ XR,
                           const unsigned char* __restrict__ mask,
                           const float* __restrict__ att, const float* __restrict__ bias,
                           const float* __restrict__ lng, const float* __restrict__ lnb,
                           const float* __restrict__ resid,
                           float* __restrict__ Hout, _Float16* __restrict__ Houth){
  __shared__ _Float16 s_xl[AN*NPROJ];                       // 22528 B
  __shared__ __align__(16) _Float16 s_xr[AN*NPROJ];         // 22528 B, reused as f32 out
  __shared__ float s_e[NHEAD*AN*AN];                        // 7744 B
  __shared__ float s_mv[AN*2];
  int g = blockIdx.x, tid = threadIdx.x;
  size_t base = (size_t)g*AN*NPROJ;
  for (int i = tid; i < AN*NPROJ; i += 256){
    s_xl[i] = (_Float16)XL[base + i];
    s_xr[i] = (_Float16)XR[base + i];
  }
  __syncthreads();
  // e[h][s][t] = sum_o lrelu(xl[s,h,o]+xr[t,h,o])*att[h,o]
  for (int p = tid; p < NHEAD*AN*AN; p += 256){
    int h = p/(AN*AN), rem = p%(AN*AN), s = rem/AN, t = rem%AN;
    const _Float16* xls = s_xl + s*NPROJ + h*HID;
    const _Float16* xrt = s_xr + t*NPROJ + h*HID;
    const float* ah = att + h*HID;
    float sum = 0.f;
    for (int o=0;o<HID;o++){
      float v = (float)xls[o] + (float)xrt[o];
      v = (v > 0.f) ? v : 0.2f*v;
      sum += v*ah[o];
    }
    s_e[p] = sum;
  }
  __syncthreads();
  // masked softmax over s for each (h,t); alpha stored in-place
  const unsigned char* mg = mask + (size_t)g*(AN*AN);
  for (int p = tid; p < NHEAD*AN; p += 256){
    int h = p/AN, t = p%AN;
    float m = -1e30f;
    for (int s=0;s<AN;s++) if (mg[s*AN+t]) { float v = s_e[h*AN*AN + s*AN + t]; m = fmaxf(m, v); }
    float sum = 0.f;
    for (int s=0;s<AN;s++){
      int idx = h*AN*AN + s*AN + t;
      float w = mg[s*AN+t] ? __expf(s_e[idx]-m) : 0.f;
      s_e[idx] = w; sum += w;
    }
    float rinv = 1.f/sum;
    for (int s=0;s<AN;s++) s_e[h*AN*AN + s*AN + t] *= rinv;
  }
  __syncthreads();
  // out[t,o] = mean_h sum_s alpha[h,s,t]*xl[s,h,o] + bias[o]  (xr LDS reused)
  float* s_out = (float*)s_xr;
  for (int p = tid; p < AN*HID; p += 256){
    int t = p/HID, o = p%HID;
    float acc = 0.f;
    for (int h=0;h<NHEAD;h++){
      const float* al = s_e + h*AN*AN;
      for (int s=0;s<AN;s++) acc += al[s*AN+t] * (float)s_xl[s*NPROJ + h*HID + o];
    }
    s_out[p] = acc*(1.f/(float)NHEAD) + bias[o];
  }
  __syncthreads();
  if (tid < AN){
    float s1=0.f, s2=0.f;
    const float* rowp = s_out + tid*HID;
    for (int o=0;o<HID;o++){ float v=rowp[o]; s1+=v; s2+=v*v; }
    float mean = s1/(float)HID;
    float var  = s2/(float)HID - mean*mean;
    s_mv[tid*2] = mean; s_mv[tid*2+1] = rsqrtf(var + EPSF);
  }
  __syncthreads();
  size_t ob = (size_t)g*AN*HID;
  for (int p = tid; p < AN*HID; p += 256){
    int t = p/HID, o = p%HID;
    float v = (s_out[p]-s_mv[t*2])*s_mv[t*2+1]*lng[o] + lnb[o];
    v = fmaxf(v, 0.f);
    if (resid) v += resid[ob + p];
    Hout[ob + p] = v;
    if (Houth) Houth[ob + p] = (_Float16)v;
  }
}

// ---------------- pre-GRU layernorm over GH=2816, emit f16 ----------------
__global__ void k_pre_ln(const float* __restrict__ X, const float* __restrict__ gg,
                         const float* __restrict__ bb, _Float16* __restrict__ Y){
  __shared__ float r1[256], r2[256];
  int row = blockIdx.x, tid = threadIdx.x;
  const float* xr = X + (size_t)row*GHD;
  float s1=0.f, s2=0.f;
  for (int c=tid; c<GHD; c+=256){ float v=xr[c]; s1+=v; s2+=v*v; }
  r1[tid]=s1; r2[tid]=s2; __syncthreads();
  for (int s=128; s>0; s>>=1){ if (tid<s){ r1[tid]+=r1[tid+s]; r2[tid]+=r2[tid+s]; } __syncthreads(); }
  float mean = r1[0]/(float)GHD;
  float var  = r2[0]/(float)GHD - mean*mean;
  float rstd = rsqrtf(var + EPSF);
  _Float16* yr = Y + (size_t)row*GHD;
  for (int c=tid; c<GHD; c+=256) yr[c] = (_Float16)((xr[c]-mean)*rstd*gg[c] + bb[c]);
}

// ---------------- GRU recurrence: one block per batch element, 128 threads ----------------
__global__ void k_gru(const float* __restrict__ gi, const float* __restrict__ Whh,
                      const float* __restrict__ bhh,
                      _Float16* __restrict__ ys_h, _Float16* __restrict__ hlast_h, int seq){
  __shared__ float sh[HID];
  int b = blockIdx.x, j = threadIdx.x;
  sh[j] = 0.f; __syncthreads();
  for (int t=0; t<seq; ++t){
    const float* girow = gi + ((size_t)(b*seq + t))*(3*HID);
    float hr = bhh[j], hz = bhh[HID+j], hn = bhh[2*HID+j];
    const float* wr = Whh + (size_t)j*HID;
    const float* wz = Whh + (size_t)(HID+j)*HID;
    const float* wn = Whh + (size_t)(2*HID+j)*HID;
    for (int k=0;k<HID;k++){ float hk = sh[k]; hr += hk*wr[k]; hz += hk*wz[k]; hn += hk*wn[k]; }
    float r = sigmoidf_(girow[j] + hr);
    float z = sigmoidf_(girow[HID+j] + hz);
    float n = tanhf(girow[2*HID+j] + r*hn);
    float h2 = (1.f - z)*n + z*sh[j];
    __syncthreads();
    sh[j] = h2;
    if (ys_h) ys_h[((size_t)(b*seq + t))*HID + j] = (_Float16)h2;
    __syncthreads();
  }
  if (hlast_h) hlast_h[(size_t)b*HID + j] = (_Float16)sh[j];
}

// ---------------- head layernorm + relu over DFF=256, emit f16 ----------------
__global__ void k_head_ln(const float* __restrict__ X, const float* __restrict__ gg,
                          const float* __restrict__ bb, _Float16* __restrict__ Y){
  __shared__ float r1[256], r2[256];
  int row = blockIdx.x, tid = threadIdx.x;
  float v = X[(size_t)row*DFFD + tid];
  r1[tid]=v; r2[tid]=v*v; __syncthreads();
  for (int s=128; s>0; s>>=1){ if (tid<s){ r1[tid]+=r1[tid+s]; r2[tid]+=r2[tid+s]; } __syncthreads(); }
  float mean = r1[0]/(float)DFFD;
  float var  = r2[0]/(float)DFFD - mean*mean;
  float rstd = rsqrtf(var + EPSF);
  float y = fmaxf((v-mean)*rstd*gg[tid] + bb[tid], 0.f);
  Y[(size_t)row*DFFD + tid] = (_Float16)y;
}

// ---------------------------------------------------------------------------
extern "C" void kernel_launch(void* const* d_in, const int* in_sizes, int n_in,
                              void* d_out, int out_size, void* d_ws, size_t ws_size,
                              hipStream_t stream){
  (void)in_sizes; (void)n_in; (void)out_size; (void)ws_size;
  const float* x_enc  = (const float*)d_in[0];
  const float* g0_Wl  = (const float*)d_in[4];
  const float* g0_bl  = (const float*)d_in[5];
  const float* g0_Wr  = (const float*)d_in[6];
  const float* g0_br  = (const float*)d_in[7];
  const float* g0_att = (const float*)d_in[8];
  const float* g0_b   = (const float*)d_in[9];
  const float* ln0_g  = (const float*)d_in[10];
  const float* ln0_b  = (const float*)d_in[11];
  const float* g1_Wl  = (const float*)d_in[12];
  const float* g1_bl  = (const float*)d_in[13];
  const float* g1_Wr  = (const float*)d_in[14];
  const float* g1_br  = (const float*)d_in[15];
  const float* g1_att = (const float*)d_in[16];
  const float* g1_b   = (const float*)d_in[17];
  const float* ln1_g  = (const float*)d_in[18];
  const float* ln1_b  = (const float*)d_in[19];
  const float* pre_g  = (const float*)d_in[20];
  const float* pre_b  = (const float*)d_in[21];
  const float* Wih0   = (const float*)d_in[22];
  const float* Whh0   = (const float*)d_in[23];
  const float* bih0   = (const float*)d_in[24];
  const float* bhh0   = (const float*)d_in[25];
  const float* Wih1   = (const float*)d_in[26];
  const float* Whh1   = (const float*)d_in[27];
  const float* bih1   = (const float*)d_in[28];
  const float* bhh1   = (const float*)d_in[29];
  const float* dW1    = (const float*)d_in[30];
  const float* db1    = (const float*)d_in[31];
  const float* dln_g  = (const float*)d_in[32];
  const float* dln_b  = (const float*)d_in[33];
  const float* dW2    = (const float*)d_in[34];
  const float* db2    = (const float*)d_in[35];

  char* base = (char*)d_ws; size_t off = 0;
  auto alloc = [&](size_t bytes)->void*{ void* r = base + off; off += (bytes + 255) & ~(size_t)255; return r; };

  unsigned char* mask = (unsigned char*)alloc((size_t)NG*AN*AN);
  float*     XL    = (float*)alloc((size_t)NROW*NPROJ*4);
  float*     XR    = (float*)alloc((size_t)NROW*NPROJ*4);
  float*     H0    = (float*)alloc((size_t)NROW*HID*4);
  _Float16*  H0h   = (_Float16*)alloc((size_t)NROW*HID*2);
  float*     H1    = (float*)alloc((size_t)NROW*HID*4);
  _Float16*  Gh    = (_Float16*)alloc((size_t)NG*GHD*2);
  float*     GI0   = (float*)alloc((size_t)NG*3*HID*4);
  _Float16*  YS0h  = (_Float16*)alloc((size_t)NG*HID*2);
  float*     GI1   = (float*)alloc((size_t)NG*3*HID*4);
  _Float16*  HLh   = (_Float16*)alloc((size_t)NBATCH*HID*2);
  float*     D1    = (float*)alloc((size_t)NBATCH*DFFD*4);
  _Float16*  D1h   = (_Float16*)alloc((size_t)NBATCH*DFFD*2);
  _Float16*  W1lT  = (_Float16*)alloc((size_t)NPROJ*HID*2);
  _Float16*  W1rT  = (_Float16*)alloc((size_t)NPROJ*HID*2);
  _Float16*  Wih0h = (_Float16*)alloc((size_t)3*HID*GHD*2);
  _Float16*  Wih1h = (_Float16*)alloc((size_t)3*HID*HID*2);
  _Float16*  dW1T  = (_Float16*)alloc((size_t)DFFD*HID*2);
  _Float16*  dW2T  = (_Float16*)alloc((size_t)NOUT*DFFD*2);

  auto cdiv = [](long a, long b)->int{ return (int)((a + b - 1)/b); };
  auto gemm_grid = [&](int M, int N)->dim3{ return dim3((unsigned)(N/NBLK), (unsigned)cdiv(M, MBLK)); };

  // weight prep
  k_transpose_f16<<<cdiv((long)HID*NPROJ,256),256,0,stream>>>(g1_Wl, W1lT, HID, NPROJ);
  k_transpose_f16<<<cdiv((long)HID*NPROJ,256),256,0,stream>>>(g1_Wr, W1rT, HID, NPROJ);
  k_transpose_f16<<<cdiv((long)HID*DFFD,256),256,0,stream>>>(dW1, dW1T, HID, DFFD);
  k_transpose_f16<<<cdiv((long)DFFD*NOUT,256),256,0,stream>>>(dW2, dW2T, DFFD, NOUT);
  k_cvt_f16<<<cdiv((long)3*HID*GHD,256),256,0,stream>>>(Wih0, Wih0h, 3*HID*GHD);
  k_cvt_f16<<<cdiv((long)3*HID*HID,256),256,0,stream>>>(Wih1, Wih1h, 3*HID*HID);

  // adjacency mask
  k_mask<<<NG, 32, 0, stream>>>(x_enc, mask);

  // GAT layer 0 (fin=4): VALU projections, then fused attention
  k_gat0_proj<<<cdiv((long)NROW*NPROJ,256),256,0,stream>>>(x_enc, g0_Wl, g0_bl, g0_Wr, g0_br, XL, XR);
  k_gat_attn<<<NG, 256, 0, stream>>>(XL, XR, mask, g0_att, g0_b, ln0_g, ln0_b, nullptr, H0, H0h);

  // GAT layer 1 (fin=128): WMMA projections (TDM-staged B), then attention w/ residual
  k_wmma_gemm<<<gemm_grid(NROW, NPROJ),256,0,stream>>>(H0h, W1lT, g1_bl, XL, NROW, NPROJ, HID);
  k_wmma_gemm<<<gemm_grid(NROW, NPROJ),256,0,stream>>>(H0h, W1rT, g1_br, XR, NROW, NPROJ, HID);
  k_gat_attn<<<NG, 256, 0, stream>>>(XL, XR, mask, g1_att, g1_b, ln1_g, ln1_b, H0, H1, nullptr);

  // pre-GRU layernorm (H1 rows are exactly the (B,S,GH) rows)
  k_pre_ln<<<NG, 256, 0, stream>>>(H1, pre_g, pre_b, Gh);

  // GRU layer 0: big input GEMM (WMMA) + recurrence
  k_wmma_gemm<<<gemm_grid(NG, 3*HID),256,0,stream>>>(Gh, Wih0h, bih0, GI0, NG, 3*HID, GHD);
  k_gru<<<NBATCH, HID, 0, stream>>>(GI0, Whh0, bhh0, YS0h, nullptr, SEQL);

  // GRU layer 1
  k_wmma_gemm<<<gemm_grid(NG, 3*HID),256,0,stream>>>(YS0h, Wih1h, bih1, GI1, NG, 3*HID, HID);
  k_gru<<<NBATCH, HID, 0, stream>>>(GI1, Whh1, bhh1, nullptr, HLh, SEQL);

  // head: d = relu(LN(hlast @ dW1 + db1)); out = d @ dW2 + db2
  k_wmma_gemm<<<gemm_grid(NBATCH, DFFD),256,0,stream>>>(HLh, dW1T, db1, D1, NBATCH, DFFD, HID);
  k_head_ln<<<NBATCH, DFFD, 0, stream>>>(D1, dln_g, dln_b, D1h);
  k_wmma_gemm<<<gemm_grid(NBATCH, NOUT),256,0,stream>>>(D1h, dW2T, db2, (float*)d_out, NBATCH, NOUT, DFFD);
}
